// LatentThoughtCondenser_9268539424921
// MI455X (gfx1250) — compile-verified, matrix operand order checked by reading, not run on hardware
//
#include <hip/hip_runtime.h>

#define LL 8
#define BB 4
#define TT 4096
#define DD 1024
#define KP 64            // number of probes
#define LN_EPS 1e-5f
#define DPAD 132         // 128 + 4 pad: 8*132 mod 64 == 32 -> lane halves hit disjoint banks

typedef __attribute__((ext_vector_type(16))) _Float16 v16h;
typedef __attribute__((ext_vector_type(8)))  float    v8f;

#if defined(__has_builtin)
#if __has_builtin(__builtin_amdgcn_global_load_async_to_lds_b128)
#define HAVE_ASYNC_LDS 1
#endif
#endif

#ifdef HAVE_ASYNC_LDS
typedef int v4i_gcc __attribute__((vector_size(16)));
typedef __attribute__((address_space(1))) v4i_gcc gas1_v4i;
typedef __attribute__((address_space(3))) v4i_gcc las3_v4i;
#define NBUF 3
#else
#define NBUF 1
#endif

// Load 16 f32 laid out as [p..p+7] and [p+16..p+23], convert to v16h.
// Matches the 16-bit WMMA A/B per-lane layout when the caller adds hi*8 to p.
__device__ __forceinline__ v16h ld_k32(const float* __restrict__ p) {
    float4 x0 = ((const float4*)p)[0];
    float4 x1 = ((const float4*)p)[1];
    float4 x2 = ((const float4*)(p + 16))[0];
    float4 x3 = ((const float4*)(p + 16))[1];
    v16h h;
    h[0]  = (_Float16)x0.x; h[1]  = (_Float16)x0.y; h[2]  = (_Float16)x0.z; h[3]  = (_Float16)x0.w;
    h[4]  = (_Float16)x1.x; h[5]  = (_Float16)x1.y; h[6]  = (_Float16)x1.z; h[7]  = (_Float16)x1.w;
    h[8]  = (_Float16)x2.x; h[9]  = (_Float16)x2.y; h[10] = (_Float16)x2.z; h[11] = (_Float16)x2.w;
    h[12] = (_Float16)x3.x; h[13] = (_Float16)x3.y; h[14] = (_Float16)x3.z; h[15] = (_Float16)x3.w;
    return h;
}

__device__ __forceinline__ float wave_max_f(float x) {
#pragma unroll
    for (int o = 16; o > 0; o >>= 1) x = fmaxf(x, __shfl_xor(x, o, 32));
    return x;
}
__device__ __forceinline__ float wave_sum_f(float x) {
#pragma unroll
    for (int o = 16; o > 0; o >>= 1) x += __shfl_xor(x, o, 32);
    return x;
}
// block of 256 threads = 8 waves
__device__ __forceinline__ float block_max_f(float x, float* sm) {
    int w = threadIdx.x >> 5, ln = threadIdx.x & 31;
    x = wave_max_f(x);
    if (ln == 0) sm[w] = x;
    __syncthreads();
    float y = (ln < 8) ? sm[ln] : -__builtin_inff();
    y = wave_max_f(y);
    if (threadIdx.x == 0) sm[0] = y;
    __syncthreads();
    float r = sm[0];
    __syncthreads();
    return r;
}
__device__ __forceinline__ float block_sum_f(float x, float* sm) {
    int w = threadIdx.x >> 5, ln = threadIdx.x & 31;
    x = wave_sum_f(x);
    if (ln == 0) sm[w] = x;
    __syncthreads();
    float y = (ln < 8) ? sm[ln] : 0.0f;
    y = wave_sum_f(y);
    if (threadIdx.x == 0) sm[0] = y;
    __syncthreads();
    float r = sm[0];
    __syncthreads();
    return r;
}

// ---------------------------------------------------------------------------
// Kernel 1: raw scores[l,b,k,t] = (probes . keys) / 32, via WMMA f16.
// grid = (T/128, L*B), block = 256 (8 waves); wave w covers 16 t-columns.
// ---------------------------------------------------------------------------
__global__ void ltc_scores_kernel(const float* __restrict__ keys,
                                  const float* __restrict__ probes,
                                  float* __restrict__ attn) {
    const int lb   = blockIdx.y;
    const int wave = threadIdx.x >> 5;
    const int lane = threadIdx.x & 31;
    const int ln   = lane & 15;
    const int hi   = lane >> 4;          // 0: K 0..7/16..23, 1: K 8..15/24..31
    const int t0   = blockIdx.x * 128 + wave * 16;

    const float* kbase = keys + (size_t)lb * TT * DD + (size_t)(t0 + ln) * DD + hi * 8;

    v8f acc[4];
#pragma unroll
    for (int m = 0; m < 4; ++m) acc[m] = (v8f){0.f,0.f,0.f,0.f,0.f,0.f,0.f,0.f};

    for (int kk = 0; kk < DD; kk += 32) {
        v16h bt = ld_k32(kbase + kk);               // B: keys^T tile (32x16)
#pragma unroll
        for (int m = 0; m < 4; ++m) {
            const float* ap = probes + (size_t)(m * 16 + ln) * DD + kk + hi * 8;
            v16h at = ld_k32(ap);                   // A: probes tile (16x32)
            acc[m] = __builtin_amdgcn_wmma_f32_16x16x32_f16(
                false, at, false, bt, (short)0, acc[m], false, false);
        }
    }

    const float scale = 0.03125f;                    // 1/sqrt(1024)
#pragma unroll
    for (int m = 0; m < 4; ++m) {
#pragma unroll
        for (int r = 0; r < 8; ++r) {
            int mg = m * 16 + r + hi * 8;            // C layout: VGPR r -> M=r(+8)
            attn[((size_t)lb * KP + mg) * TT + t0 + ln] = acc[m][r] * scale;
        }
    }
}

// ---------------------------------------------------------------------------
// Kernel 2: in-place masked softmax over each row of 4096.
// grid = L*B*K rows, block = 256; each thread owns 16 elements.
// ---------------------------------------------------------------------------
__global__ void ltc_softmax_kernel(float* __restrict__ attn,
                                   const int* __restrict__ amask) {
    __shared__ float sm[8];
    const int row = blockIdx.x;                      // lb*K + k
    const int lb  = row >> 6;
    const int b   = lb & (BB - 1);
    float* p = attn + (size_t)row * TT;
    const int* mrow = amask + (size_t)b * TT;

    float v[16];
    float mx = -__builtin_inff();
#pragma unroll
    for (int i = 0; i < 16; ++i) {
        int t = i * 256 + threadIdx.x;
        float s = p[t];
        s = (mrow[t] == 0) ? -__builtin_inff() : s;
        v[i] = s;
        mx = fmaxf(mx, s);
    }
    mx = block_max_f(mx, sm);

    float sum = 0.f;
#pragma unroll
    for (int i = 0; i < 16; ++i) {
        v[i] = __expf(v[i] - mx);
        sum += v[i];
    }
    sum = block_sum_f(sum, sm);
    float inv = 1.0f / sum;
#pragma unroll
    for (int i = 0; i < 16; ++i) {
        p[i * 256 + threadIdx.x] = v[i] * inv;
    }
}

// ---------------------------------------------------------------------------
// Kernel 3: ck = attn x keys, cv = attn x values (raw, pre-layernorm).
// grid = (D/128, L*B), block = 256 (8 waves); wave w covers 16 d-columns.
// K/V tiles (32 t x 128 d, f32) staged in LDS via async global->LDS copies;
// TRIPLE buffered with two tiles in flight per block (ASYNCcnt deep pipeline)
// to cover HBM latency at low block counts. B operands are built with
// bank-clean strided ds reads + f16 convert.
// ---------------------------------------------------------------------------
__global__ void ltc_context_kernel(const float* __restrict__ keys,
                                   const float* __restrict__ values,
                                   const float* __restrict__ attn,
                                   float* __restrict__ ck,
                                   float* __restrict__ cv) {
    __shared__ float kbuf[NBUF][32][DPAD];
    __shared__ float vbuf[NBUF][32][DPAD];

    const int lb   = blockIdx.y;
    const int tid  = threadIdx.x;
    const int wave = tid >> 5;
    const int lane = tid & 31;
    const int ln   = lane & 15;
    const int hi   = lane >> 4;
    const int dblk = blockIdx.x * 128;
    const int dloc = wave * 16 + ln;

    const float* kb = keys   + (size_t)lb * TT * DD + dblk;
    const float* vb = values + (size_t)lb * TT * DD + dblk;

    // tile-copy mapping: thread -> (row, 16 consecutive floats)
    const int crow = tid >> 3;           // 0..31
    const int cd0  = (tid & 7) * 16;     // 0..112

    v8f ak[4], av[4];
#pragma unroll
    for (int m = 0; m < 4; ++m) {
        ak[m] = (v8f){0.f,0.f,0.f,0.f,0.f,0.f,0.f,0.f};
        av[m] = (v8f){0.f,0.f,0.f,0.f,0.f,0.f,0.f,0.f};
    }

    auto compute = [&](int buf, int tglob) {
        const float* kp = &kbuf[buf][0][0];
        const float* vp = &vbuf[buf][0][0];
        v16h bk, bv;
#pragma unroll
        for (int j = 0; j < 8; ++j) {
            int r0 = hi * 8 + j;
            int r1 = r0 + 16;
            bk[j]     = (_Float16)kp[r0 * DPAD + dloc];
            bk[j + 8] = (_Float16)kp[r1 * DPAD + dloc];
            bv[j]     = (_Float16)vp[r0 * DPAD + dloc];
            bv[j + 8] = (_Float16)vp[r1 * DPAD + dloc];
        }
#pragma unroll
        for (int m = 0; m < 4; ++m) {
            const float* ap = attn + ((size_t)lb * KP + m * 16 + ln) * TT + tglob + hi * 8;
            v16h at = ld_k32(ap);                   // A: attn tile (16x32)
            ak[m] = __builtin_amdgcn_wmma_f32_16x16x32_f16(
                false, at, false, bk, (short)0, ak[m], false, false);
            av[m] = __builtin_amdgcn_wmma_f32_16x16x32_f16(
                false, at, false, bv, (short)0, av[m], false, false);
        }
    };

#ifdef HAVE_ASYNC_LDS
    auto issue = [&](int buf, int t) {
        const float* gk = kb + (size_t)(t + crow) * DD + cd0;
        const float* gv = vb + (size_t)(t + crow) * DD + cd0;
#pragma unroll
        for (int i = 0; i < 4; ++i) {
            __builtin_amdgcn_global_load_async_to_lds_b128(
                (gas1_v4i*)(gk + 4 * i), (las3_v4i*)&kbuf[buf][crow][cd0 + 4 * i], 0, 0);
            __builtin_amdgcn_global_load_async_to_lds_b128(
                (gas1_v4i*)(gv + 4 * i), (las3_v4i*)&vbuf[buf][crow][cd0 + 4 * i], 0, 0);
        }
    };
    const int steps = TT / 32;
    issue(0, 0);
    issue(1, 32);
    for (int it = 0; it < steps; ++it) {
        const int buf = it % 3;
        if (it + 2 < steps) issue((it + 2) % 3, (it + 2) * 32);
        const int rem = steps - 1 - it;              // future tiles in flight (<=2)
        // 8 async B128s per tile per thread; in-order completion per wave.
        if (rem >= 2)      asm volatile("s_wait_asynccnt 0x10" ::: "memory");
        else if (rem == 1) asm volatile("s_wait_asynccnt 0x8" ::: "memory");
        else               asm volatile("s_wait_asynccnt 0x0" ::: "memory");
        __syncthreads();                 // tile `it` fully resident for all waves
        compute(buf, it * 32);
        __syncthreads();                 // done reading before buf is re-issued
    }
#else
    // Fallback: register-double-buffered synchronous staging, single LDS buffer.
    float4 rk[4], rv[4];
    auto fetch = [&](int t) {
#pragma unroll
        for (int i = 0; i < 4; ++i) {
            rk[i] = *(const float4*)(kb + (size_t)(t + crow) * DD + cd0 + 4 * i);
            rv[i] = *(const float4*)(vb + (size_t)(t + crow) * DD + cd0 + 4 * i);
        }
    };
    fetch(0);
    for (int it = 0; it < TT / 32; ++it) {
#pragma unroll
        for (int i = 0; i < 4; ++i) {
            *(float4*)&kbuf[0][crow][cd0 + 4 * i] = rk[i];
            *(float4*)&vbuf[0][crow][cd0 + 4 * i] = rv[i];
        }
        __syncthreads();                 // staging visible to all waves
        if (it + 1 < TT / 32) fetch((it + 1) * 32);   // overlap with compute
        compute(0, it * 32);
        __syncthreads();                 // done reading before next store
    }
#endif

#pragma unroll
    for (int m = 0; m < 4; ++m) {
#pragma unroll
        for (int r = 0; r < 8; ++r) {
            int mg = m * 16 + r + hi * 8;
            ck[((size_t)lb * KP + mg) * DD + dblk + dloc] = ak[m][r];
            cv[((size_t)lb * KP + mg) * DD + dblk + dloc] = av[m][r];
        }
    }
}

// ---------------------------------------------------------------------------
// Kernel 4: in-place LayerNorm of ck and cv rows with per-layer gamma/beta.
// grid = L*B*K rows, block = 256; each thread owns 4 consecutive elements.
// ---------------------------------------------------------------------------
__global__ void ltc_lnorm_kernel(float* __restrict__ ck, float* __restrict__ cv,
                                 const float* __restrict__ kg, const float* __restrict__ kb,
                                 const float* __restrict__ vg, const float* __restrict__ vb) {
    __shared__ float sm[8];
    const int row = blockIdx.x;                      // ((l*B + b)*K + k)
    const int l   = row >> 8;                        // B*K = 256
    const int d0  = threadIdx.x * 4;

#pragma unroll
    for (int which = 0; which < 2; ++which) {
        float* p         = (which ? cv : ck) + (size_t)row * DD;
        const float* g   = (which ? vg : kg) + (size_t)l * DD;
        const float* be  = (which ? vb : kb) + (size_t)l * DD;

        float4 x = *(const float4*)(p + d0);
        float s  = x.x + x.y + x.z + x.w;
        float s2 = x.x * x.x + x.y * x.y + x.z * x.z + x.w * x.w;
        s  = block_sum_f(s, sm);
        s2 = block_sum_f(s2, sm);
        float mean = s * (1.0f / DD);
        float var  = s2 * (1.0f / DD) - mean * mean;
        float rinv = rsqrtf(var + LN_EPS);

        float4 gg = *(const float4*)(g + d0);
        float4 bb = *(const float4*)(be + d0);
        float4 o;
        o.x = (x.x - mean) * rinv * gg.x + bb.x;
        o.y = (x.y - mean) * rinv * gg.y + bb.y;
        o.z = (x.z - mean) * rinv * gg.z + bb.z;
        o.w = (x.w - mean) * rinv * gg.w + bb.w;
        *(float4*)(p + d0) = o;
    }
}

extern "C" void kernel_launch(void* const* d_in, const int* in_sizes, int n_in,
                              void* d_out, int out_size, void* d_ws, size_t ws_size,
                              hipStream_t stream) {
    const float* keys   = (const float*)d_in[0];
    const float* values = (const float*)d_in[1];
    const int*   amask  = (const int*)d_in[2];
    const float* probes = (const float*)d_in[3];
    const float* kg     = (const float*)d_in[4];
    const float* kb     = (const float*)d_in[5];
    const float* vg     = (const float*)d_in[6];
    const float* vb     = (const float*)d_in[7];

    float* out  = (float*)d_out;
    float* ck   = out;
    float* cv   = out + (size_t)LL * BB * KP * DD;
    float* attn = cv  + (size_t)LL * BB * KP * DD;

    dim3 g1(TT / 128, LL * BB);
    ltc_scores_kernel<<<g1, 256, 0, stream>>>(keys, probes, attn);

    ltc_softmax_kernel<<<LL * BB * KP, 256, 0, stream>>>(attn, amask);

    dim3 g3(DD / 128, LL * BB);
    ltc_context_kernel<<<g3, 256, 0, stream>>>(keys, values, attn, ck, cv);

    ltc_lnorm_kernel<<<LL * BB * KP, 256, 0, stream>>>(ck, cv, kg, kb, vg, vb);

    (void)in_sizes; (void)n_in; (void)out_size; (void)d_ws; (void)ws_size;
}
